// Decoder_41463614275920
// MI455X (gfx1250) — matrix operand order repaired and checked by reading, not compiled, not run on hardware
//
#include <hip/hip_runtime.h>
#include <hip/hip_bf16.h>
#include <cmath>

#define B_   32
#define T_   64
#define V_   32000
#define E_   300
#define EP_  320      // E padded to multiple of 32
#define H_   512
#define G4_  2048     // 4*H

typedef __attribute__((ext_vector_type(16))) __bf16 v16bf;
typedef __attribute__((ext_vector_type(8)))  __bf16 v8bf;
typedef __attribute__((ext_vector_type(8)))  float  v8f;

// ---------------- WMMA fragment helpers (wave32, 16x16x32 bf16) ----------------
// A matrix 16x32 (MxK), 16-bit: lanes 0-15 -> M=lane, K chunks {0..7, 16..23};
//                               lanes 16-31 -> M=lane-16, K chunks {8..15, 24..31}.
__device__ __forceinline__ v16bf load_a_bf16(const __bf16* p, int ld, int row0, int k0) {
    int lane = threadIdx.x & 31, l2 = lane & 15, hi = lane >> 4;
    const __bf16* base = p + (size_t)(row0 + l2) * ld + k0 + hi * 8;
    v8bf c0 = *(const v8bf*)(base);
    v8bf c1 = *(const v8bf*)(base + 16);
    v16bf r;
#pragma unroll
    for (int i = 0; i < 8; ++i) { r[i] = c0[i]; r[8 + i] = c1[i]; }
    return r;
}

// Same layout but converting an fp32 source on the fly (for recurrent h).
__device__ __forceinline__ v16bf load_a_f32(const float* p, int ld, int row0, int k0) {
    int lane = threadIdx.x & 31, l2 = lane & 15, hi = lane >> 4;
    const float* base = p + (size_t)(row0 + l2) * ld + k0 + hi * 8;
    v16bf r;
#pragma unroll
    for (int i = 0; i < 8; ++i) { r[i] = (__bf16)base[i]; r[8 + i] = (__bf16)base[16 + i]; }
    return r;
}

// B matrix 32x16 (KxN): lanes 0-15 -> N=lane, K chunks {0..7, 16..23}; lanes 16-31 -> N=lane-16, K chunks {8..15, 24..31}.
// Weight stored row-major [N, K], so K is contiguous per N-row.
__device__ __forceinline__ v16bf load_b_bf16(const __bf16* p, int ld, int col0, int k0) {
    int lane = threadIdx.x & 31, l2 = lane & 15, hi = lane >> 4;
    const __bf16* base = p + (size_t)(col0 + l2) * ld + k0 + hi * 8;
    v8bf c0 = *(const v8bf*)(base);
    v8bf c1 = *(const v8bf*)(base + 16);
    v16bf r;
#pragma unroll
    for (int i = 0; i < 8; ++i) { r[i] = c0[i]; r[8 + i] = c1[i]; }
    return r;
}

__device__ __forceinline__ v8f wmma_bf16(v16bf a, v16bf b, v8f c) {
    return __builtin_amdgcn_wmma_f32_16x16x32_bf16(false, a, false, b, (short)0, c, false, false);
}

__device__ __forceinline__ float sigm(float x) { return 1.0f / (1.0f + __expf(-x)); }

// ---------------- Kernel 1: h0 = init_h@Wh^T+bh ; c0 = init_c@Wc^T+bc ----------------
__global__ void k_init_hc(const float* __restrict__ init_h, const float* __restrict__ init_c,
                          const float* __restrict__ Wh, const float* __restrict__ bh,
                          const float* __restrict__ Wc, const float* __restrict__ bc,
                          float* __restrict__ h0, float* __restrict__ c0) {
    int id = blockIdx.x * blockDim.x + threadIdx.x;   // [0, 2*B*H)
    int which = id / (B_ * H_);
    int r = id % (B_ * H_);
    int b = r / H_, j = r % H_;
    const float* x = which ? init_c : init_h;
    const float* W = which ? Wc : Wh;
    const float* bi = which ? bc : bh;
    float acc = bi[j];
    const float* xp = x + (size_t)b * H_;
    const float* wp = W + (size_t)j * H_;
#pragma unroll 4
    for (int u = 0; u < H_; ++u) acc += xp[u] * wp[u];
    (which ? c0 : h0)[r] = acc;
}

// ---------------- Kernel 2: gather embedding -> bf16 A_emb [T*B, EP] ----------------
__global__ void k_gather_emb(const int* __restrict__ trg, const float* __restrict__ emb,
                             __bf16* __restrict__ A) {
    long total = (long)T_ * B_ * EP_;
    for (long i = (long)blockIdx.x * blockDim.x + threadIdx.x; i < total;
         i += (long)gridDim.x * blockDim.x) {
        int r = (int)(i / EP_), k = (int)(i % EP_);
        int t = r / B_, b = r % B_;
        int tok = trg[b * T_ + t];
        A[i] = (k < E_) ? (__bf16)emb[(size_t)tok * E_ + k] : (__bf16)0.0f;
    }
}

// ---------------- Kernel 3: fp32 [N,K] -> bf16 [N,Kpad] with zero pad ----------------
__global__ void k_cvt_pad(const float* __restrict__ src, __bf16* __restrict__ dst,
                          int N, int K, int Kpad) {
    long total = (long)N * Kpad;
    for (long i = (long)blockIdx.x * blockDim.x + threadIdx.x; i < total;
         i += (long)gridDim.x * blockDim.x) {
        int n = (int)(i / Kpad), k = (int)(i % Kpad);
        dst[i] = (k < K) ? (__bf16)src[(size_t)n * K + k] : (__bf16)0.0f;
    }
}

// ---------------- Kernel 4: x_gates[T*B, 4H] = A_emb @ W_ih^T + b_ih + b_hh --------
// block = 128 threads (4 waves); block tile M=32, N=128; wave tile 16x64.
__global__ void k_gemm_xg(const __bf16* __restrict__ A, const __bf16* __restrict__ W,
                          const float* __restrict__ bih, const float* __restrict__ bhh,
                          float* __restrict__ C) {
    int wave = threadIdx.x >> 5;
    int m0 = blockIdx.y * 32 + 16 * (wave & 1);
    int n0 = blockIdx.x * 128 + 64 * (wave >> 1);
    v8f acc[4] = {};
    for (int k = 0; k < EP_; k += 32) {
        v16bf a = load_a_bf16(A, EP_, m0, k);
#pragma unroll
        for (int j = 0; j < 4; ++j) {
            v16bf bf = load_b_bf16(W, EP_, n0 + 16 * j, k);
            acc[j] = wmma_bf16(a, bf, acc[j]);
        }
    }
    int lane = threadIdx.x & 31, l2 = lane & 15, hi = lane >> 4;
#pragma unroll
    for (int j = 0; j < 4; ++j) {
        int col = n0 + 16 * j + l2;
        float bias = bih[col] + bhh[col];
#pragma unroll
        for (int r = 0; r < 8; ++r) {
            int row = m0 + (hi ? 8 : 0) + r;
            C[(size_t)row * G4_ + col] = acc[j][r] + bias;
        }
    }
}

// ---------------- Kernel 5: one LSTM step -----------------------------------------
// block = 256 threads (8 waves); covers all B=32 rows and a 32-wide j slice of all
// four gates. wave w: m-tile = w&1, gate = w>>1 (i,f,g,o), each wave 16x32.
__global__ void k_lstm_step(const float* __restrict__ xg_t, const __bf16* __restrict__ Whh,
                            const float* __restrict__ h_in, float* __restrict__ h_out,
                            float* __restrict__ c) {
    __shared__ float lds[B_][128];   // 32 rows x (4 gates x 32 cols)
    int wave = threadIdx.x >> 5;
    int m0 = 16 * (wave & 1);
    int gate = wave >> 1;
    int j0 = blockIdx.x * 32;
    int nb = gate * H_ + j0;
    v8f acc[2] = {};
    for (int k = 0; k < H_; k += 32) {
        v16bf a = load_a_f32(h_in, H_, m0, k);
#pragma unroll
        for (int s = 0; s < 2; ++s) {
            v16bf bf = load_b_bf16(Whh, H_, nb + 16 * s, k);
            acc[s] = wmma_bf16(a, bf, acc[s]);
        }
    }
    int lane = threadIdx.x & 31, l2 = lane & 15, hi = lane >> 4;
#pragma unroll
    for (int s = 0; s < 2; ++s) {
        int col = nb + 16 * s + l2;         // column in [0, 4H)
        int lcol = gate * 32 + 16 * s + l2; // column in LDS
#pragma unroll
        for (int r = 0; r < 8; ++r) {
            int row = m0 + (hi ? 8 : 0) + r;
            lds[row][lcol] = acc[s][r] + xg_t[(size_t)row * G4_ + col];
        }
    }
    __syncthreads();
    for (int idx = threadIdx.x; idx < B_ * 32; idx += blockDim.x) {
        int brow = idx >> 5, jl = idx & 31, j = j0 + jl;
        float iv = sigm(lds[brow][jl]);
        float fv = sigm(lds[brow][32 + jl]);
        float gv = tanhf(lds[brow][64 + jl]);
        float ov = sigm(lds[brow][96 + jl]);
        size_t o = (size_t)brow * H_ + j;
        float cn = fv * c[o] + iv * gv;
        c[o] = cn;
        h_out[o] = ov * tanhf(cn);
    }
}

// ---------------- Kernel 6: A_hs = bf16(tanh(hs)) ---------------------------------
__global__ void k_tanh_cvt(const float* __restrict__ hs, __bf16* __restrict__ A) {
    long total = (long)T_ * B_ * H_;
    for (long i = (long)blockIdx.x * blockDim.x + threadIdx.x; i < total;
         i += (long)gridDim.x * blockDim.x)
        A[i] = (__bf16)tanhf(hs[i]);
}

// ---------------- Kernel 7: logits = A_hs @ W_logit^T + b_logit -------------------
// block = 256 threads (8 waves); block tile M=32, N=256; wave tile 16x64.
// The 32x512 bf16 A slab (32 KB) is staged once per block into LDS with
// GLOBAL_LOAD_ASYNC_TO_LDS_B128 (ASYNCcnt), then read via ds_load per k-step.
// hs row r = t*B + b ; output layout [B, T, V].
__global__ void k_logits(const __bf16* __restrict__ A, const __bf16* __restrict__ W,
                         const float* __restrict__ bias, float* __restrict__ out) {
    __shared__ __bf16 As[B_ * H_];   // 32 KB, at LDS offset 0 (sole shared object)
    int tid = threadIdx.x;
    int mbase = blockIdx.y * 32;

    // Cooperative async stage: 32768 B / (256 thr * 16 B) = 8 rounds.
    const char* gbase = (const char*)(A + (size_t)mbase * H_);
#pragma unroll
    for (int r = 0; r < 8; ++r) {
        unsigned boff = (unsigned)(r * 256 + tid) * 16u;   // LDS byte offset == slab offset
        const char* g = gbase + boff;
        asm volatile("global_load_async_to_lds_b128 %0, %1, off"
                     :: "v"(boff), "v"(g) : "memory");
    }
    asm volatile("s_wait_asynccnt 0x0" ::: "memory");
    __syncthreads();

    int wave = tid >> 5;
    int mlocal = 16 * (wave & 1);
    int n0 = blockIdx.x * 256 + 64 * (wave >> 1);
    v8f acc[4] = {};
    for (int k = 0; k < H_; k += 32) {
        v16bf a = load_a_bf16(As, H_, mlocal, k);          // LDS reads (ds_load)
        if (k + 32 < H_)
            __builtin_prefetch(W + (size_t)n0 * H_ + k + 32, 0, 2);
#pragma unroll
        for (int j = 0; j < 4; ++j) {
            v16bf bf = load_b_bf16(W, H_, n0 + 16 * j, k);
            acc[j] = wmma_bf16(a, bf, acc[j]);
        }
    }
    int lane = tid & 31, l2 = lane & 15, hi = lane >> 4;
#pragma unroll
    for (int j = 0; j < 4; ++j) {
        int col = n0 + 16 * j + l2;
        float bs = bias[col];
#pragma unroll
        for (int r = 0; r < 8; ++r) {
            int row = mbase + mlocal + (hi ? 8 : 0) + r;
            int t = row >> 5, b = row & 31;   // row = t*B + b, B=32
            out[((size_t)b * T_ + t) * V_ + col] = acc[j][r] + bs;
        }
    }
}

// ---------------- host-side orchestration ------------------------------------------
static inline size_t align256(size_t x) { return (x + 255) & ~(size_t)255; }

extern "C" void kernel_launch(void* const* d_in, const int* in_sizes, int n_in,
                              void* d_out, int out_size, void* d_ws, size_t ws_size,
                              hipStream_t stream) {
    const int*   trg      = (const int*)  d_in[0];
    const float* init_h   = (const float*)d_in[1];
    const float* init_c   = (const float*)d_in[2];
    const float* embedding= (const float*)d_in[3];
    const float* Wh       = (const float*)d_in[4];
    const float* bh       = (const float*)d_in[5];
    const float* Wc       = (const float*)d_in[6];
    const float* bc       = (const float*)d_in[7];
    const float* W_ih     = (const float*)d_in[8];
    const float* W_hh     = (const float*)d_in[9];
    const float* b_ih     = (const float*)d_in[10];
    const float* b_hh     = (const float*)d_in[11];
    const float* W_logit  = (const float*)d_in[12];
    const float* b_logit  = (const float*)d_in[13];
    float* out = (float*)d_out;

    char* ws = (char*)d_ws;
    size_t off = 0;
    float*  xg      = (float*) (ws + off); off = align256(off + (size_t)T_ * B_ * G4_ * 4);
    float*  hs      = (float*) (ws + off); off = align256(off + (size_t)T_ * B_ * H_ * 4);
    float*  h0      = (float*) (ws + off); off = align256(off + (size_t)B_ * H_ * 4);
    float*  cst     = (float*) (ws + off); off = align256(off + (size_t)B_ * H_ * 4);
    __bf16* A_emb   = (__bf16*)(ws + off); off = align256(off + (size_t)T_ * B_ * EP_ * 2);
    __bf16* Wih_bf  = (__bf16*)(ws + off); off = align256(off + (size_t)G4_ * EP_ * 2);
    __bf16* Whh_bf  = (__bf16*)(ws + off); off = align256(off + (size_t)G4_ * H_ * 2);
    __bf16* Wlog_bf = (__bf16*)(ws + off); off = align256(off + (size_t)V_ * H_ * 2);
    __bf16* A_hs    = (__bf16*)(ws + off); off = align256(off + (size_t)T_ * B_ * H_ * 2);
    (void)ws_size; (void)in_sizes; (void)n_in; (void)out_size;

    // 1. initial hidden/cell state
    k_init_hc<<<dim3((2 * B_ * H_) / 256), dim3(256), 0, stream>>>(
        init_h, init_c, Wh, bh, Wc, bc, h0, cst);

    // 2. gather + convert embedding rows, convert weights to bf16
    k_gather_emb<<<dim3(2560), dim3(256), 0, stream>>>(trg, embedding, A_emb);
    k_cvt_pad<<<dim3(2560), dim3(256), 0, stream>>>(W_ih, Wih_bf, G4_, E_, EP_);
    k_cvt_pad<<<dim3(4096), dim3(256), 0, stream>>>(W_hh, Whh_bf, G4_, H_, H_);
    k_cvt_pad<<<dim3(8192), dim3(256), 0, stream>>>(W_logit, Wlog_bf, V_, H_, H_);

    // 3. x_gates GEMM (includes both biases)
    k_gemm_xg<<<dim3(G4_ / 128, (T_ * B_) / 32), dim3(128), 0, stream>>>(
        A_emb, Wih_bf, b_ih, b_hh, xg);

    // 4. serial LSTM recurrence; h double-buffered through hs
    for (int t = 0; t < T_; ++t) {
        const float* h_in = (t == 0) ? h0 : (hs + (size_t)(t - 1) * B_ * H_);
        float* h_out = hs + (size_t)t * B_ * H_;
        const float* xg_t = xg + (size_t)t * B_ * G4_;
        k_lstm_step<<<dim3(H_ / 32), dim3(256), 0, stream>>>(xg_t, Whh_bf, h_in, h_out, cst);
    }

    // 5. tanh + convert, then the big logit GEMM
    k_tanh_cvt<<<dim3(4096), dim3(256), 0, stream>>>(hs, A_hs);
    k_logits<<<dim3(V_ / 256, (T_ * B_) / 32), dim3(256), 0, stream>>>(
        A_hs, Wlog_bf, b_logit, out);
}
